// SwinTransformerBlock_12816182411474
// MI455X (gfx1250) — matrix operand order repaired and checked by reading, not compiled, not run on hardware
//
#include <hip/hip_runtime.h>

// ---------------- problem constants ----------------
// B=32, H=W=56, C=192, WS=7, SS=3, NH=6, N=49(pad 64), NW=64, HD=32, HID=768
// windows = B*NW = 2048 ; tokens = B*H*W = 100352

#define DEVFN static __device__ __forceinline__

typedef __attribute__((ext_vector_type(16))) __bf16 v16bf;
typedef __attribute__((ext_vector_type(8)))  __bf16 v8bf;
typedef __attribute__((ext_vector_type(8)))  float  v8f;

#if defined(__HIP_DEVICE_COMPILE__) && defined(__gfx1250__) && \
    __has_builtin(__builtin_amdgcn_global_load_async_to_lds_b32)
#define HAVE_ASYNC_LDS 1
typedef __attribute__((address_space(1))) int* as1_i32p;
typedef __attribute__((address_space(3))) int* as3_i32p;
#endif

DEVFN unsigned short f2bf(float f) {            // f32 -> bf16 RNE
  unsigned u = __builtin_bit_cast(unsigned, f);
  u += 0x7FFFu + ((u >> 16) & 1u);
  return (unsigned short)(u >> 16);
}
DEVFN int div7(int v) { return (v * 147) >> 10; }   // exact for 0..208
DEVFN v8f vzero() { v8f z = {0.f,0.f,0.f,0.f,0.f,0.f,0.f,0.f}; return z; }

DEVFN v8f wmma_bf16(v16bf a, v16bf b, v8f c) {
  return __builtin_amdgcn_wmma_f32_16x16x32_bf16(false, a, false, b, (short)0, c,
                                                 false, false);
}

DEVFN void wait_async_lds() {
#ifdef HAVE_ASYNC_LDS
#if __has_builtin(__builtin_amdgcn_s_wait_asynccnt)
  __builtin_amdgcn_s_wait_asynccnt(0);
#else
  asm volatile("s_wait_asynccnt 0x0" ::: "memory");
#endif
#endif
}

// A fragment (16x32) from row-major bf16 [row][k], lda in elements.
// low lanes hold K 0..15, high lanes K 16..31, row = lane%16.
DEVFN v16bf frag_a(const unsigned short* base, int row0, int lda, int k0, int lane) {
  const int r = lane & 15, hi = (lane >> 4) & 1;
  const __bf16* p = (const __bf16*)base + (size_t)(row0 + r) * lda + k0 + hi * 16;
  v8bf lo = *(const v8bf*)p;
  v8bf hv = *(const v8bf*)(p + 8);
  v16bf o;
#pragma unroll
  for (int i = 0; i < 8; i++) { o[i] = lo[i]; o[i + 8] = hv[i]; }
  return o;
}

// B fragment (32x16) where storage holds B^T row-major: elem(k,n) = base[(n0+n)*ldn + k].
// Same order as frag_a: low lanes K 0..15, high lanes K 16..31, n = lane%16 -> 16 contiguous elems/lane.
DEVFN v16bf frag_bt(const unsigned short* base, int n0, int ldn, int k0, int lane) {
  const int n = lane & 15, hi = (lane >> 4) & 1;
  const __bf16* p = (const __bf16*)base + (size_t)(n0 + n) * ldn + k0 + hi * 16;
  v8bf lo = *(const v8bf*)p;
  v8bf hv = *(const v8bf*)(p + 8);
  v16bf o;
#pragma unroll
  for (int i = 0; i < 8; i++) { o[i] = lo[i]; o[i + 8] = hv[i]; }
  return o;
}

// Pre-swizzled weight fragment: 512 bf16 per tile, lane loads 16 contiguous (32B).
DEVFN v16bf frag_w(const unsigned short* wf, int tile, int lane) {
  return *(const v16bf*)((const __bf16*)wf + (size_t)tile * 512 + lane * 16);
}

// ---------------- kernel 1: weight fragment pack ----------------
__global__ void prep_weights(const float* __restrict__ src, unsigned short* __restrict__ dst,
                             int K, int Ncols) {
  const int NT = Ncols >> 4;
  const int tiles = (K >> 5) * NT;
  for (int tile = blockIdx.x; tile < tiles; tile += gridDim.x) {
    const int kt = tile / NT, nt = tile % NT;
    for (int e = threadIdx.x; e < 512; e += blockDim.x) {
      const int lane = e >> 4, i = e & 15;
      const int kl = ((lane >> 4) << 4) + i;   // (lane>=16?16:0)+i
      const int nl = lane & 15;
      dst[(size_t)tile * 512 + e] =
          f2bf(src[(size_t)(kt * 32 + kl) * Ncols + nt * 16 + nl]);
    }
  }
}

// ---------------- kernel 2: fused LN1+shift+QKV+attention+proj+residual ----------------
// LDS layout (dynamic):
static constexpr int XLN_LD = 200;   // [64][200] bf16 : LN'd tokens, later attn-out concat
static constexpr int QKV_LD = 392;   // [64][392] bf16 : q(0..191) k(192..383); later P(0..383)
static constexpr int VT_LD  = 72;    // [192][72] bf16 : V transposed (rows = h*32+d, cols = j)
static constexpr int OFF_QKV = 64 * XLN_LD * 2;            // 25600
static constexpr int OFF_VT  = OFF_QKV + 64 * QKV_LD * 2;  // 75776
static constexpr int OFF_RPB = OFF_VT + 192 * VT_LD * 2;   // 103424  (1014 f32 used)
static constexpr int OFF_RED = OFF_RPB + 4064;             // 107488  (2 x 512 f32)
static constexpr int OFF_IDW = OFF_RED + 4096;             // 111584  (64 int)
static constexpr int SMEM_A  = OFF_IDW + 256;              // 111840 bytes

__global__ __launch_bounds__(384, 1)
void swin_attn(const float* __restrict__ x, const float* __restrict__ g1,
               const float* __restrict__ b1, const unsigned short* __restrict__ qkvw,
               const float* __restrict__ rpb_table, const unsigned short* __restrict__ projw,
               float* __restrict__ hbuf) {
  extern __shared__ char smem[];
  unsigned short* XLN = (unsigned short*)smem;
  unsigned short* QKV = (unsigned short*)(smem + OFF_QKV);
  unsigned short* VT  = (unsigned short*)(smem + OFF_VT);
  float* RPB  = (float*)(smem + OFF_RPB);
  float* REDS = (float*)(smem + OFF_RED);
  float* REDQ = REDS + 512;
  int*   IDW  = (int*)(smem + OFF_IDW);

  const int tid  = threadIdx.x;
  const int lane = tid & 31;
  const int wave = tid >> 5;           // 12 waves
  const int wid  = blockIdx.x;         // 0..2047
  const int bb   = wid >> 6;
  const int w    = wid & 63;
  const int wy   = w >> 3, wx = w & 7;

  // stage relative-position-bias table (169*6 f32) -- async LDS DMA when available
#ifdef HAVE_ASYNC_LDS
  for (int i = tid; i < 1014; i += 384) {
    __builtin_amdgcn_global_load_async_to_lds_b32(
        (as1_i32p)(rpb_table + i), (as3_i32p)(RPB + i), 0, 0);
  }
#else
  for (int i = tid; i < 1014; i += 384) RPB[i] = rpb_table[i];
#endif

  // ---- LN1 + cyclic-shift gather into XLN (bf16), region ids for mask ----
  {
    const int row = tid / 6;           // 0..63
    const int t6  = tid % 6;
    const int c0  = t6 * 32;
    float vals[32];
    if (row < 49) {
      const int ty = div7(row), tx = row - 7 * ty;
      const int sy = wy * 7 + ty, sx = wx * 7 + tx;       // shifted-image coords
      int ysrc = sy + 3; if (ysrc >= 56) ysrc -= 56;      // roll(-3) gather
      int xsrc = sx + 3; if (xsrc >= 56) xsrc -= 56;
      const size_t srcbase = ((size_t)bb * 3136 + ysrc * 56 + xsrc) * 192;
      float s = 0.f, q = 0.f;
#pragma unroll
      for (int i = 0; i < 32; i++) {
        float v = x[srcbase + c0 + i];
        vals[i] = v; s += v; q += v * v;
      }
      REDS[row * 8 + t6] = s; REDQ[row * 8 + t6] = q;
      if (t6 == 0) {
        const int ry = (sy < 49) ? 0 : ((sy < 53) ? 1 : 2);
        const int rx = (sx < 49) ? 0 : ((sx < 53) ? 1 : 2);
        IDW[row] = ry * 3 + rx;
      }
    } else if (t6 == 0) {
      IDW[row] = -1;
    }
    wait_async_lds();                  // RPB resident before anyone reads it
    __syncthreads();
    if (row < 49) {
      float s = 0.f, q = 0.f;
#pragma unroll
      for (int i = 0; i < 6; i++) { s += REDS[row * 8 + i]; q += REDQ[row * 8 + i]; }
      const float mu = s * (1.f / 192.f);
      const float var = q * (1.f / 192.f) - mu * mu;
      const float rstd = rsqrtf(var + 1e-5f);
#pragma unroll
      for (int i = 0; i < 32; i++) {
        const int c = c0 + i;
        XLN[row * XLN_LD + c] = f2bf((vals[i] - mu) * rstd * g1[c] + b1[c]);
      }
    } else {
#pragma unroll
      for (int i = 0; i < 32; i++) XLN[row * XLN_LD + c0 + i] = 0;
    }
  }
  __syncthreads();   // barrier 1: XLN ready

  const int nn = lane & 15, hi = (lane >> 4) & 1;

  // ---- QKV GEMM: [64x192] x [192x576] ; q scaled by HD^-0.5 ; v stored transposed ----
  for (int t = wave; t < 144; t += 12) {
    const int mt = t / 36, nt = t % 36;
    __builtin_prefetch((const __bf16*)qkvw + (size_t)nt * 512, 0, 0);
    v8f acc = vzero();
#pragma unroll
    for (int k = 0; k < 6; k++) {
      v16bf a  = frag_a(XLN, mt * 16, XLN_LD, k * 32, lane);
      v16bf bw = frag_w(qkvw, k * 36 + nt, lane);
      acc = wmma_bf16(a, bw, acc);
    }
    const int ncol = nt * 16 + nn;
    if (nt < 24) {                       // q (scaled) and k, row-major [token][col]
      const float sc = (nt < 12) ? 0.17677669529663687f : 1.f;
#pragma unroll
      for (int vg = 0; vg < 8; vg++)
        QKV[(mt * 16 + vg + 8 * hi) * QKV_LD + ncol] = f2bf(acc[vg] * sc);
    } else {                             // v transposed: VT[h*32+d][token]
      const int vr = ncol - 384;
#pragma unroll
      for (int vg = 0; vg < 8; vg++)
        VT[vr * VT_LD + (mt * 16 + vg + 8 * hi)] = f2bf(acc[vg]);
    }
  }
  __syncthreads();   // barrier 2: q/k/VT ready

  // ---- attention: wave -> (head, half of i-tiles) ----
  const int h = wave >> 1;        // 0..5
  const int halfw = wave & 1;     // i-tiles {2*halfw, 2*halfw+1}
  v8f sacc[2][4];

  // S = (q*scale) @ k^T  -- HD=32 => single WMMA per 16x16 tile
#pragma unroll
  for (int ii = 0; ii < 2; ii++) {
    const int it = halfw * 2 + ii;
    v16bf aq = frag_a(QKV + h * 32, it * 16, QKV_LD, 0, lane);
#pragma unroll
    for (int jt = 0; jt < 4; jt++) {
      v16bf bk = frag_bt(QKV + 192 + h * 32, jt * 16, QKV_LD, 0, lane);
      sacc[ii][jt] = wmma_bf16(aq, bk, vzero());
    }
  }

  // hoisted per-column (j) mask data: invariant across ii/vg
  int jy4[4], jx4[4], idj4[4], jok4[4];
#pragma unroll
  for (int jt = 0; jt < 4; jt++) {
    const int j = jt * 16 + nn;
    jok4[jt] = (j < 49);
    const int jy = div7(j);
    jy4[jt] = jy; jx4[jt] = j - 7 * jy;
    idj4[jt] = IDW[j];
  }

  // bias (rel-pos + region mask) and softmax, all in registers
#pragma unroll
  for (int ii = 0; ii < 2; ii++) {
    const int it = halfw * 2 + ii;
#pragma unroll
    for (int vg = 0; vg < 8; vg++) {
      const int i = it * 16 + vg + 8 * hi;
      const int iok = (i < 49);
      const int iy = div7(i), ix = i - 7 * iy;
      const int idi = IDW[i];
#pragma unroll
      for (int jt = 0; jt < 4; jt++) {
        float s = -1e30f;
        if (iok && jok4[jt]) {
          float bias = RPB[((ix - jx4[jt] + 6) * 13 + (iy - jy4[jt] + 6)) * 6 + h];
          if (idi == idj4[jt]) bias -= 100.f;   // faithful to reference quirk
          s = sacc[ii][jt][vg] + bias;
        }
        sacc[ii][jt][vg] = s;
      }
    }
#pragma unroll
    for (int vg = 0; vg < 8; vg++) {
      float m = sacc[ii][0][vg];
#pragma unroll
      for (int jt = 1; jt < 4; jt++) m = fmaxf(m, sacc[ii][jt][vg]);
      m = fmaxf(m, __shfl_xor(m, 1, 32));
      m = fmaxf(m, __shfl_xor(m, 2, 32));
      m = fmaxf(m, __shfl_xor(m, 4, 32));
      m = fmaxf(m, __shfl_xor(m, 8, 32));
      float s = 0.f;
#pragma unroll
      for (int jt = 0; jt < 4; jt++) {
        float p = __expf(sacc[ii][jt][vg] - m);
        sacc[ii][jt][vg] = p;
        s += p;
      }
      s += __shfl_xor(s, 1, 32);
      s += __shfl_xor(s, 2, 32);
      s += __shfl_xor(s, 4, 32);
      s += __shfl_xor(s, 8, 32);
      const float rs = 1.f / s;
#pragma unroll
      for (int jt = 0; jt < 4; jt++) sacc[ii][jt][vg] *= rs;
    }
  }
  __syncthreads();   // barrier 3: all q/k reads done, P may overwrite cols 0..383

  // store P (bf16) over the dead q/k region: P[h] at col h*64
#pragma unroll
  for (int ii = 0; ii < 2; ii++) {
    const int it = halfw * 2 + ii;
#pragma unroll
    for (int jt = 0; jt < 4; jt++)
#pragma unroll
      for (int vg = 0; vg < 8; vg++)
        QKV[(it * 16 + vg + 8 * hi) * QKV_LD + h * 64 + jt * 16 + nn] =
            f2bf(sacc[ii][jt][vg]);
  }

  // out = P @ V  (K=64 -> 2 steps), write concat [64][192] into XLN region
#pragma unroll
  for (int ii = 0; ii < 2; ii++) {
    const int it = halfw * 2 + ii;
#pragma unroll
    for (int dt = 0; dt < 2; dt++) {
      v8f o = vzero();
#pragma unroll
      for (int ks = 0; ks < 2; ks++) {
        v16bf a  = frag_a(QKV + h * 64, it * 16, QKV_LD, ks * 32, lane);
        v16bf bv = frag_bt(VT + (size_t)(h * 32) * VT_LD, dt * 16, VT_LD, ks * 32, lane);
        o = wmma_bf16(a, bv, o);
      }
#pragma unroll
      for (int vg = 0; vg < 8; vg++)
        XLN[(it * 16 + vg + 8 * hi) * XLN_LD + h * 32 + dt * 16 + nn] = f2bf(o[vg]);
    }
  }
  __syncthreads();   // barrier 4: attn-out concat ready

  // ---- proj GEMM [64x192] x [192x192] + window-reverse + unroll + residual -> hbuf ----
  for (int t = wave; t < 48; t += 12) {
    const int mt = t & 3, nt = t >> 2;     // nt 0..11
    v8f acc = vzero();
#pragma unroll
    for (int k = 0; k < 6; k++) {
      v16bf a  = frag_a(XLN, mt * 16, XLN_LD, k * 32, lane);
      v16bf bw = frag_w(projw, k * 12 + nt, lane);
      acc = wmma_bf16(a, bw, acc);
    }
#pragma unroll
    for (int vg = 0; vg < 8; vg++) {
      const int i = mt * 16 + vg + 8 * hi;
      if (i < 49) {
        const int ty = div7(i), tx = i - 7 * ty;
        const int sy = wy * 7 + ty, sx = wx * 7 + tx;
        int y = sy + 3; if (y >= 56) y -= 56;   // roll(+3) scatter == gather mapping
        int xc = sx + 3; if (xc >= 56) xc -= 56;
        const size_t idx = ((size_t)bb * 3136 + y * 56 + xc) * 192 + nt * 16 + nn;
        hbuf[idx] = x[idx] + acc[vg];
      }
    }
  }
}

// ---------------- kernel 3: fused LN2 + FC1 + GELU + FC2 + residual ----------------
static constexpr int HID_LD  = 776;                         // [64][776] bf16
static constexpr int OFF_HID = 64 * XLN_LD * 2;             // 25600
static constexpr int OFF_REDM = OFF_HID + 64 * HID_LD * 2;  // 124928
static constexpr int SMEM_M  = OFF_REDM + 4096;             // 129024 bytes

__global__ __launch_bounds__(256, 1)
void swin_mlp(const float* __restrict__ hbuf, const float* __restrict__ g2,
              const float* __restrict__ b2, const unsigned short* __restrict__ fc1w,
              const unsigned short* __restrict__ fc2w, float* __restrict__ out) {
  extern __shared__ char smem[];
  unsigned short* XLN = (unsigned short*)smem;
  unsigned short* HID = (unsigned short*)(smem + OFF_HID);
  float* REDS = (float*)(smem + OFF_REDM);
  float* REDQ = REDS + 512;

  const int tid = threadIdx.x, lane = tid & 31, wave = tid >> 5;   // 8 waves
  const int r0 = blockIdx.x * 64;

  // ---- LN2 into XLN (bf16) ----
  {
    const int row = tid >> 2, t4 = tid & 3, c0 = t4 * 48;
    float vals[48];
    float s = 0.f, q = 0.f;
    const size_t base = (size_t)(r0 + row) * 192;
#pragma unroll
    for (int i = 0; i < 48; i++) {
      float v = hbuf[base + c0 + i];
      vals[i] = v; s += v; q += v * v;
    }
    REDS[row * 8 + t4] = s; REDQ[row * 8 + t4] = q;
    __syncthreads();
    s = 0.f; q = 0.f;
#pragma unroll
    for (int i = 0; i < 4; i++) { s += REDS[row * 8 + i]; q += REDQ[row * 8 + i]; }
    const float mu = s * (1.f / 192.f);
    const float var = q * (1.f / 192.f) - mu * mu;
    const float rstd = rsqrtf(var + 1e-5f);
#pragma unroll
    for (int i = 0; i < 48; i++) {
      const int c = c0 + i;
      XLN[row * XLN_LD + c] = f2bf((vals[i] - mu) * rstd * g2[c] + b2[c]);
    }
  }
  __syncthreads();

  const int nn = lane & 15, hi = (lane >> 4) & 1;

  // ---- FC1 [64x192]x[192x768] + tanh-GELU -> HID (bf16, stays in LDS) ----
  for (int t = wave; t < 192; t += 8) {
    const int mt = t & 3, nt = t >> 2;     // nt 0..47
    __builtin_prefetch((const __bf16*)fc1w + (size_t)nt * 512, 0, 0);
    v8f acc = vzero();
#pragma unroll
    for (int k = 0; k < 6; k++) {
      v16bf a  = frag_a(XLN, mt * 16, XLN_LD, k * 32, lane);
      v16bf bw = frag_w(fc1w, k * 48 + nt, lane);
      acc = wmma_bf16(a, bw, acc);
    }
#pragma unroll
    for (int vg = 0; vg < 8; vg++) {
      const float xg = acc[vg];
      const float g = 0.5f * xg * (1.f + tanhf(0.7978845608f * (xg + 0.044715f * xg * xg * xg)));
      HID[(mt * 16 + vg + 8 * hi) * HID_LD + nt * 16 + nn] = f2bf(g);
    }
  }
  __syncthreads();

  // ---- FC2 [64x768]x[768x192] + residual -> out ----
  for (int t = wave; t < 48; t += 8) {
    const int mt = t & 3, nt = t >> 2;     // nt 0..11
    v8f acc = vzero();
#pragma unroll
    for (int k = 0; k < 24; k++) {
      v16bf a  = frag_a(HID, mt * 16, HID_LD, k * 32, lane);
      v16bf bw = frag_w(fc2w, k * 12 + nt, lane);
      acc = wmma_bf16(a, bw, acc);
    }
#pragma unroll
    for (int vg = 0; vg < 8; vg++) {
      const int i = mt * 16 + vg + 8 * hi;
      const size_t idx = (size_t)(r0 + i) * 192 + nt * 16 + nn;
      out[idx] = hbuf[idx] + acc[vg];
    }
  }
}

// ---------------- host launcher ----------------
extern "C" void kernel_launch(void* const* d_in, const int* in_sizes, int n_in,
                              void* d_out, int out_size, void* d_ws, size_t ws_size,
                              hipStream_t stream) {
  (void)in_sizes; (void)n_in; (void)out_size; (void)ws_size;
  const float* x      = (const float*)d_in[0];
  const float* g1     = (const float*)d_in[1];
  const float* b1     = (const float*)d_in[2];
  const float* qkv_w  = (const float*)d_in[3];
  const float* rpb    = (const float*)d_in[4];
  const float* proj_w = (const float*)d_in[5];
  const float* g2     = (const float*)d_in[6];
  const float* b2     = (const float*)d_in[7];
  const float* fc1_w  = (const float*)d_in[8];
  const float* fc2_w  = (const float*)d_in[9];
  float* out = (float*)d_out;

  char* ws = (char*)d_ws;
  size_t off = 0;
  auto take = [&](size_t bytes) {
    char* p = ws + off;
    off = (off + bytes + 255) & ~(size_t)255;
    return p;
  };
  unsigned short* qkvw_f = (unsigned short*)take(216 * 512 * 2);   // 192x576
  unsigned short* projw_f = (unsigned short*)take(72 * 512 * 2);   // 192x192
  unsigned short* fc1w_f = (unsigned short*)take(288 * 512 * 2);   // 192x768
  unsigned short* fc2w_f = (unsigned short*)take(288 * 512 * 2);   // 768x192
  float* hbuf = (float*)take((size_t)32 * 3136 * 192 * 4);         // residual stream

  prep_weights<<<216, 128, 0, stream>>>(qkv_w, qkvw_f, 192, 576);
  prep_weights<<<72, 128, 0, stream>>>(proj_w, projw_f, 192, 192);
  prep_weights<<<288, 128, 0, stream>>>(fc1_w, fc1w_f, 192, 768);
  prep_weights<<<288, 128, 0, stream>>>(fc2_w, fc2w_f, 768, 192);

  swin_attn<<<2048, 384, SMEM_A, stream>>>(x, g1, b1, qkvw_f, rpb, projw_f, hbuf);
  swin_mlp<<<1568, 256, SMEM_M, stream>>>(hbuf, g2, b2, fc1w_f, fc2w_f, out);
}
